// ConvAttLIF_86887188399013
// MI455X (gfx1250) — matrix operand order repaired and checked by reading, not compiled, not run on hardware
//
#include <hip/hip_runtime.h>
#include <math.h>

typedef __attribute__((ext_vector_type(16))) _Float16 v16h;
typedef __attribute__((ext_vector_type(8)))  _Float16 v8h;
typedef __attribute__((ext_vector_type(8)))  float    v8f;
typedef int v4i __attribute__((vector_size(16)));   // matches async-builtin param

#define Bn     16
#define Tn     20
#define CINn   64
#define Hn     32
#define Wn     32
#define CHn    128
#define HWn    1024
#define KTOT   576          // CIN*3*3
#define FRAMES 320          // B*T
#define NCHUNK 18           // KTOT/32
#define LDSTR  48           // padded halfs per LDS row (96B, 16B aligned)

#define AS1 __attribute__((address_space(1)))
#define AS3 __attribute__((address_space(3)))

#if __has_builtin(__builtin_amdgcn_global_load_async_to_lds_b128) && \
    __has_builtin(__builtin_amdgcn_s_wait_asynccnt)
#define HAVE_ASYNC 1
#else
#define HAVE_ASYNC 0
#endif

// ---------------------------------------------------------------------------
// Kernel 0: one-time fp32 -> fp16 weight conversion into workspace.
// ---------------------------------------------------------------------------
__global__ __launch_bounds__(256)
void wconv_kernel(const float* __restrict__ cw, _Float16* __restrict__ wq)
{
    int i = blockIdx.x * 256 + threadIdx.x;
    if (i < CHn * KTOT) wq[i] = (_Float16)cw[i];
}

// ---------------------------------------------------------------------------
// Kernel 1: implicit-GEMM 3x3 conv via v_wmma_f32_16x16x32_f16.
// 256 threads (8 waves); block tile M=128 x N=128 pixels of one frame.
// Double-buffered LDS; A chunk via async global->LDS copy when available,
// B chunk register-staged so its global loads overlap the WMMAs.
// ---------------------------------------------------------------------------
__global__ __launch_bounds__(256)
void conv_wmma_kernel(const float* __restrict__ data,
                      const _Float16* __restrict__ wq,
                      const float* __restrict__ cb,
                      float* __restrict__ out,
                      float* __restrict__ sumP,
                      float* __restrict__ maxP)
{
    __shared__ __align__(16) _Float16 Alds[2][CHn * LDSTR];   // weights [c][k]
    __shared__ __align__(16) _Float16 Blds[2][128 * LDSTR];   // im2col  [n][k]
    __shared__ float redS[8];
    __shared__ float redM[8];

    const int frame = blockIdx.y;            // 0..319 (= b*T+t)
    const int p0    = blockIdx.x * 128;      // pixel tile base
    const int tid   = threadIdx.x;
    const int wave  = tid >> 5;
    const int lane  = tid & 31;
    const int wm    = wave >> 2;             // 0..1  -> M offset wm*64
    const int wn    = wave & 3;              // 0..3  -> N offset wn*32
    const int half  = lane >> 4;             // lane half (ISA frag layout)
    const int l16   = lane & 15;

    const float* fdata = data + (size_t)frame * (CINn * HWn);

    // A staging: thread owns weight row ac, 16-half group ahb (32B per chunk)
    const int ac = tid >> 1, ahb = tid & 1;
    const _Float16* aG = wq + ac * KTOT + ahb * 16;
    const int aLoff = ac * LDSTR + ahb * 16;

    // B staging: thread owns pixel n, contiguous k-range bh*16..bh*16+15
    const int bh = tid >> 7, bn = tid & 127;
    const int p  = p0 + bn;
    const int ph = p >> 5, pw = p & 31;

    v8f acc[4][2] = {};

    // ---- prologue: stage chunk 0 into buffer 0 ----
    {
        v8h a0 = *(const v8h*)(aG);
        v8h a1 = *(const v8h*)(aG + 8);
        *(v8h*)&Alds[0][aLoff]     = a0;
        *(v8h*)&Alds[0][aLoff + 8] = a1;
        v8h x0, x1;
        #pragma unroll
        for (int i = 0; i < 16; ++i) {
            int gk = bh * 16 + i;
            int ci = gk / 9, rs = gk % 9;
            int r  = rs / 3, s  = rs % 3;
            int ih = ph + r - 1, iw = pw + s - 1;
            float v = ((unsigned)ih < 32u && (unsigned)iw < 32u)
                          ? fdata[ci * HWn + ih * Wn + iw] : 0.f;
            if (i < 8) x0[i] = (_Float16)v; else x1[i - 8] = (_Float16)v;
        }
        *(v8h*)&Blds[0][bn * LDSTR + bh * 16]     = x0;
        *(v8h*)&Blds[0][bn * LDSTR + bh * 16 + 8] = x1;
        __syncthreads();
    }

    for (int c = 0; c < NCHUNK; ++c) {
        const int cur = c & 1, nxt = cur ^ 1;
        const bool more = (c + 1 < NCHUNK);
        const int kc2 = (c + 1) * 32;

        // ---- issue next-chunk loads (in flight during WMMA compute) ----
        v8h a0, a1;
        float bv[16];
        if (more) {
#if HAVE_ASYNC
            __builtin_amdgcn_global_load_async_to_lds_b128(
                (AS1 v4i*)(aG + kc2),
                (AS3 v4i*)&Alds[nxt][aLoff], 0, 0);
            __builtin_amdgcn_global_load_async_to_lds_b128(
                (AS1 v4i*)(aG + kc2 + 8),
                (AS3 v4i*)&Alds[nxt][aLoff + 8], 0, 0);
#else
            a0 = *(const v8h*)(aG + kc2);
            a1 = *(const v8h*)(aG + kc2 + 8);
#endif
            #pragma unroll
            for (int i = 0; i < 16; ++i) {
                int gk = kc2 + bh * 16 + i;
                int ci = gk / 9, rs = gk % 9;
                int r  = rs / 3, s  = rs % 3;
                int ih = ph + r - 1, iw = pw + s - 1;
                bv[i] = ((unsigned)ih < 32u && (unsigned)iw < 32u)
                            ? fdata[ci * HWn + ih * Wn + iw] : 0.f;
            }
        }

        // ---- fragment loads, contiguous k = 16*half + s in both A and B + 8 WMMAs ----
        union U { v16h v; v8h h[2]; };
        v16h af[4], bf[2];
        #pragma unroll
        for (int mt = 0; mt < 4; ++mt) {
            int row = wm * 64 + mt * 16 + l16;
            U u;
            u.h[0] = *(const v8h*)&Alds[cur][row * LDSTR + half * 16];     // K kb..+7
            u.h[1] = *(const v8h*)&Alds[cur][row * LDSTR + half * 16 + 8]; // K kb+8..+15
            af[mt] = u.v;
        }
        #pragma unroll
        for (int nt = 0; nt < 2; ++nt) {
            int col = wn * 32 + nt * 16 + l16;
            U u;
            u.h[0] = *(const v8h*)&Blds[cur][col * LDSTR + half * 16];     // K kb..+7
            u.h[1] = *(const v8h*)&Blds[cur][col * LDSTR + half * 16 + 8]; // K kb+8..+15
            bf[nt] = u.v;
        }
        #pragma unroll
        for (int mt = 0; mt < 4; ++mt)
            #pragma unroll
            for (int nt = 0; nt < 2; ++nt)
                acc[mt][nt] = __builtin_amdgcn_wmma_f32_16x16x32_f16(
                    false, af[mt], false, bf[nt],
                    (short)0, acc[mt][nt], false, false);

        // ---- commit staged next chunk ----
        if (more) {
#if !HAVE_ASYNC
            *(v8h*)&Alds[nxt][aLoff]     = a0;
            *(v8h*)&Alds[nxt][aLoff + 8] = a1;
#endif
            v8h x0, x1;
            #pragma unroll
            for (int i = 0; i < 16; ++i) {
                if (i < 8) x0[i] = (_Float16)bv[i]; else x1[i - 8] = (_Float16)bv[i];
            }
            *(v8h*)&Blds[nxt][bn * LDSTR + bh * 16]     = x0;
            *(v8h*)&Blds[nxt][bn * LDSTR + bh * 16 + 8] = x1;
#if HAVE_ASYNC
            __builtin_amdgcn_s_wait_asynccnt(0);
#endif
        }
        __syncthreads();
    }

    // ---- epilogue: bias add, store y, block sum/max partials ----
    float lsum = 0.f, lmax = -3.0e38f;
    float* fout = out + (size_t)frame * (CHn * HWn);
    #pragma unroll
    for (int mt = 0; mt < 4; ++mt) {
        #pragma unroll
        for (int nt = 0; nt < 2; ++nt) {
            int n = p0 + wn * 32 + nt * 16 + l16;
            #pragma unroll
            for (int v = 0; v < 8; ++v) {
                int m = v + half * 8;               // C/D layout: VGPR v -> M
                int cc = wm * 64 + mt * 16 + m;
                float val = acc[mt][nt][v] + cb[cc];
                fout[cc * HWn + n] = val;
                lsum += val;
                lmax  = fmaxf(lmax, val);
            }
        }
    }
    #pragma unroll
    for (int off = 16; off > 0; off >>= 1) {
        lsum += __shfl_xor(lsum, off, 32);
        lmax  = fmaxf(lmax, __shfl_xor(lmax, off, 32));
    }
    if (lane == 0) { redS[wave] = lsum; redM[wave] = lmax; }
    __syncthreads();
    if (tid == 0) {
        float s = 0.f, m = -3.0e38f;
        #pragma unroll
        for (int i = 0; i < 8; ++i) { s += redS[i]; m = fmaxf(m, redM[i]); }
        sumP[frame * 8 + blockIdx.x] = s;    // deterministic partials, no atomics
        maxP[frame * 8 + blockIdx.x] = m;
    }
}

// ---------------------------------------------------------------------------
// Kernel 2: temporal attention (tiny, deterministic partial fold + MLP).
// ---------------------------------------------------------------------------
__global__ void att_kernel(const float* __restrict__ sumP,
                           const float* __restrict__ maxP,
                           const float* __restrict__ w1,   // [5][20]
                           const float* __restrict__ w2,   // [20][5]
                           float* __restrict__ att)
{
    int b = threadIdx.x;
    if (b >= Bn) return;
    float avg[Tn], mx[Tn];
    for (int t = 0; t < Tn; ++t) {
        int f = b * Tn + t;
        float s = 0.f, m = -3.0e38f;
        for (int i = 0; i < 8; ++i) {
            s += sumP[f * 8 + i];
            m  = fmaxf(m, maxP[f * 8 + i]);
        }
        avg[t] = s * (1.0f / (float)(CHn * HWn));
        mx[t]  = m;
    }
    float ha[5], hm[5];
    for (int j = 0; j < 5; ++j) {
        float sa = 0.f, sm = 0.f;
        for (int t = 0; t < Tn; ++t) {
            float wv = w1[j * Tn + t];
            sa += avg[t] * wv;
            sm += mx[t]  * wv;
        }
        ha[j] = fmaxf(sa, 0.f);
        hm[j] = fmaxf(sm, 0.f);
    }
    for (int t = 0; t < Tn; ++t) {
        float s = 0.f;
        for (int j = 0; j < 5; ++j)
            s += (ha[j] + hm[j]) * w2[t * 5 + j];
        att[b * Tn + t] = 1.f / (1.f + __expf(-s));
    }
}

// ---------------------------------------------------------------------------
// Kernel 3: LIF scan over T, in place on d_out, float4-vectorized.
// ---------------------------------------------------------------------------
__global__ __launch_bounds__(256)
void lif_kernel(float4* __restrict__ y, const float* __restrict__ att)
{
    int idx = blockIdx.x * 256 + threadIdx.x;   // 0 .. B*CH*HW/4 - 1
    int b   = idx >> 15;                        // / (CH*HW/4 = 32768)
    int rem = idx & 32767;
    size_t base = (size_t)b * (Tn * CHn * HWn / 4) + rem;
    const float* attb = att + b * Tn;
    float4 h = make_float4(0.f, 0.f, 0.f, 0.f);
    #pragma unroll
    for (int t = 0; t < Tn; ++t) {
        size_t a = base + (size_t)t * (CHn * HWn / 4);
        float4 x = y[a];
        float av = attb[t];
        float4 r;
#define LIF1(c) { float u = h.c * 0.3f + x.c * av;                    \
                  float sp = (u >= 0.6f) ? 1.f : 0.f;                 \
                  r.c = sp; h.c = u * (1.f - sp); }
        LIF1(x) LIF1(y) LIF1(z) LIF1(w)
#undef LIF1
        y[a] = r;
    }
}

extern "C" void kernel_launch(void* const* d_in, const int* in_sizes, int n_in,
                              void* d_out, int out_size, void* d_ws, size_t ws_size,
                              hipStream_t stream)
{
    const float* data = (const float*)d_in[0];
    const float* cw   = (const float*)d_in[1];
    const float* cb   = (const float*)d_in[2];
    const float* w1   = (const float*)d_in[3];
    const float* w2   = (const float*)d_in[4];
    float* out = (float*)d_out;

    float*    sumP = (float*)d_ws;                    // [320*8]
    float*    maxP = sumP + FRAMES * 8;               // [320*8]
    float*    att  = maxP + FRAMES * 8;               // [320]
    _Float16* wq   = (_Float16*)(att + FRAMES);       // [128*576] f16 (16B aligned)

    wconv_kernel<<<(CHn * KTOT + 255) / 256, 256, 0, stream>>>(cw, wq);
    dim3 grid(8, FRAMES);
    conv_wmma_kernel<<<grid, 256, 0, stream>>>(data, wq, cb, out, sumP, maxP);
    att_kernel<<<1, 32, 0, stream>>>(sumP, maxP, w1, w2, att);
    lif_kernel<<<(Bn * CHn * HWn / 4) / 256, 256, 0, stream>>>((float4*)out, att);
}
